// DelphiM4_3547642987217
// MI455X (gfx1250) — matrix operand order repaired and checked by reading, not compiled, not run on hardware
//
#include <hip/hip_runtime.h>
#include <hip/hip_bf16.h>
#include <math.h>

// ---------------------------------------------------------------- types
typedef __bf16 bf16_t;
typedef __attribute__((ext_vector_type(16))) __bf16 v16bf;
typedef __attribute__((ext_vector_type(8)))  __bf16 v8bf;
typedef __attribute__((ext_vector_type(8)))  float  v8f;

#define N_EMBD   768
#define N_HEAD   12
#define HEAD_DIM 64
#define BATCH    4
#define LM       1024
#define LS       1024
#define HALFN    512
#define FFN      3072

union FragAB { v16bf v; v8bf h[2]; };

__device__ __forceinline__ v8f wmma_bf16(const FragAB& a, const FragAB& b, v8f c) {
  return __builtin_amdgcn_wmma_f32_16x16x32_bf16(false, a.v, false, b.v,
                                                 (short)0, c, false, false);
}

// ------------------------------------------------- fuse_embed: occurrence scan
__global__ void occ_kernel(const int* __restrict__ mod_idx, int* __restrict__ entry) {
  if (threadIdx.x != 0) return;
  int b = blockIdx.x;
  int c2 = 0, c3 = 0;
  for (int i = 0; i < LS; ++i) {
    int idx = mod_idx[b * LS + i];
    if (idx == 2) {
      int r = c2 < HALFN - 1 ? c2 : HALFN - 1;
      entry[b * LS + i] = (b * HALFN + r);
      ++c2;
    } else {
      int r = c3 < HALFN - 1 ? c3 : HALFN - 1;
      entry[b * LS + i] = 0x40000000 | (b * HALFN + r);
      ++c3;
    }
  }
}

__global__ void gather_kernel(const int* __restrict__ entry,
                              const float* __restrict__ m2,
                              const float* __restrict__ m3,
                              float* __restrict__ modx) {
  int row = blockIdx.x;                 // b*LS + i
  int e = entry[row];
  const float* src = (e & 0x40000000) ? m3 : m2;
  int r = e & 0x3FFFFFFF;
  for (int t = threadIdx.x; t < N_EMBD; t += blockDim.x)
    modx[(size_t)row * N_EMBD + t] = src[(size_t)r * N_EMBD + t];
}

// ---------------------------------------------------------------- LayerNorm -> bf16
__global__ __launch_bounds__(256) void ln_kernel(const float* __restrict__ in,
                                                 const float* __restrict__ w,
                                                 const float* __restrict__ bia,
                                                 bf16_t* __restrict__ out) {
  int row = blockIdx.x;
  const float* x = in + (size_t)row * N_EMBD;
  int t = threadIdx.x;
  float v0 = x[t], v1 = x[t + 256], v2 = x[t + 512];
  float s  = v0 + v1 + v2;
  float ss = v0 * v0 + v1 * v1 + v2 * v2;
  #pragma unroll
  for (int d = 1; d < 32; d <<= 1) { s += __shfl_xor(s, d); ss += __shfl_xor(ss, d); }
  __shared__ float sh[2][8];
  int wid = t >> 5, lane = t & 31;
  if (lane == 0) { sh[0][wid] = s; sh[1][wid] = ss; }
  __syncthreads();
  s = sh[0][lane & 7]; ss = sh[1][lane & 7];
  #pragma unroll
  for (int d = 1; d < 8; d <<= 1) { s += __shfl_xor(s, d); ss += __shfl_xor(ss, d); }
  float mean = s * (1.0f / N_EMBD);
  float var  = ss * (1.0f / N_EMBD) - mean * mean;
  float rst  = rsqrtf(var + 1e-5f);
  bf16_t* o = out + (size_t)row * N_EMBD;
  o[t]       = (bf16_t)((v0 - mean) * rst * w[t]       + bia[t]);
  o[t + 256] = (bf16_t)((v1 - mean) * rst * w[t + 256] + bia[t + 256]);
  o[t + 512] = (bf16_t)((v2 - mean) * rst * w[t + 512] + bia[t + 512]);
}

// ---------------------------------------------------------------- f32 -> bf16
__global__ void cvt_kernel(const float* __restrict__ src, bf16_t* __restrict__ dst, int n) {
  int i = blockIdx.x * blockDim.x + threadIdx.x;
  if (i < n) dst[i] = (bf16_t)src[i];
}

// ---------------------------------------------------------------- GEMM  C = A @ W^T + bias
// A: [M,K] bf16 row-major, W: [N,K] bf16 row-major (B-fragment pattern: lane=n, K contiguous)
// MODE 0: bf16 out   1: bf16 out scattered to Vt[b,h,d,l]   2: f32 out + residual   3: bf16 GELU
template <int MODE>
__global__ __launch_bounds__(256) void gemm_kernel(
    const bf16_t* __restrict__ A, const bf16_t* __restrict__ Bw,
    const float* __restrict__ bias, const float* __restrict__ resid,
    float* __restrict__ outF, bf16_t* __restrict__ outH,
    int M, int N, int K, int tilesN) {
  int blk = blockIdx.x;
  int bm = blk / tilesN, bn = blk % tilesN;
  int tid = threadIdx.x;
  int wid = tid >> 5, lane = tid & 31;
  int wm = wid >> 2, wn = wid & 3;
  int half = lane >> 4, l15 = lane & 15;
  int mBase = bm * 64 + wm * 32;
  int nBase = bn * 128 + wn * 32;

  v8f acc[2][2];
  #pragma unroll
  for (int fn = 0; fn < 2; ++fn) {
    float bv = bias[nBase + fn * 16 + l15];
    #pragma unroll
    for (int fm = 0; fm < 2; ++fm)
      #pragma unroll
      for (int v = 0; v < 8; ++v) acc[fm][fn][v] = bv;
  }
  const int kh = half * 8;   // A lane sub-offset
  const int kb = half * 8;   // B lane sub-offset
  for (int k0 = 0; k0 < K; k0 += 32) {
    FragAB af[2], bfr[2];
    #pragma unroll
    for (int fm = 0; fm < 2; ++fm) {
      const bf16_t* ap = A + (size_t)(mBase + fm * 16 + l15) * K + k0 + kh;
      af[fm].h[0] = *(const v8bf*)ap;
      af[fm].h[1] = *(const v8bf*)(ap + 16);
    }
    #pragma unroll
    for (int fn = 0; fn < 2; ++fn) {
      const bf16_t* bp = Bw + (size_t)(nBase + fn * 16 + l15) * K + k0 + kb;
      bfr[fn].h[0] = *(const v8bf*)bp;
      bfr[fn].h[1] = *(const v8bf*)(bp + 16);
    }
    #pragma unroll
    for (int fm = 0; fm < 2; ++fm)
      #pragma unroll
      for (int fn = 0; fn < 2; ++fn)
        acc[fm][fn] = wmma_bf16(af[fm], bfr[fn], acc[fm][fn]);
  }
  #pragma unroll
  for (int fm = 0; fm < 2; ++fm)
    #pragma unroll
    for (int fn = 0; fn < 2; ++fn)
      #pragma unroll
      for (int v = 0; v < 8; ++v) {
        int m = mBase + fm * 16 + v + half * 8;
        int n = nBase + fn * 16 + l15;
        float val = acc[fm][fn][v];
        if (MODE == 0) {
          outH[(size_t)m * N + n] = (bf16_t)val;
        } else if (MODE == 1) {
          int b = m >> 10, l = m & 1023, hh = n >> 6, d = n & 63;
          outH[((size_t)((b * N_HEAD + hh) * HEAD_DIM + d) << 10) | l] = (bf16_t)val;
        } else if (MODE == 2) {
          outF[(size_t)m * N + n] = resid[(size_t)m * N + n] + val;
        } else {
          float g = 0.5f * val * (1.0f + erff(val * 0.70710678118f));
          outH[(size_t)m * N + n] = (bf16_t)g;
        }
      }
}

// ---------------------------------------------------------------- flash cross-attention
// one wave = 16 query rows of one (b,h); streams 1024 keys in 32-chunks
__global__ __launch_bounds__(256) void attn_kernel(
    const bf16_t* __restrict__ qb, const bf16_t* __restrict__ kb,
    const bf16_t* __restrict__ vt, const float* __restrict__ age,
    const float* __restrict__ mod_age, bf16_t* __restrict__ yb) {
  __shared__ bf16_t plds[8][16 * 32];
  int tid = threadIdx.x;
  int wid = tid >> 5, lane = tid & 31;
  int half = lane >> 4, l15 = lane & 15;
  int wg = blockIdx.x * 8 + wid;
  int b = wg / (N_HEAD * 64);
  int r = wg % (N_HEAD * 64);
  int h = r / 64;
  int q0 = (r % 64) * 16;

  float arow[8];
  #pragma unroll
  for (int v = 0; v < 8; ++v) arow[v] = age[b * LM + q0 + v + half * 8];

  FragAB aq[2];
  {
    const bf16_t* qp = qb + (size_t)(b * LM + q0 + l15) * N_EMBD + h * HEAD_DIM;
    #pragma unroll
    for (int c = 0; c < 2; ++c) {
      aq[c].h[0] = *(const v8bf*)(qp + c * 32 + half * 8);
      aq[c].h[1] = *(const v8bf*)(qp + c * 32 + 16 + half * 8);
    }
  }

  float mi[8], li[8];
  v8f acc[4];
  #pragma unroll
  for (int v = 0; v < 8; ++v) { mi[v] = -1e30f; li[v] = 0.0f; }
  #pragma unroll
  for (int j = 0; j < 4; ++j)
    #pragma unroll
    for (int v = 0; v < 8; ++v) acc[j][v] = 0.0f;

  for (int kc = 0; kc < LS; kc += 32) {
    // --- S = Q K^T for 32 keys (2 N-frags, 2 d-chunks each)
    v8f s[2];
    #pragma unroll
    for (int t = 0; t < 2; ++t) {
      #pragma unroll
      for (int v = 0; v < 8; ++v) s[t][v] = 0.0f;
      const bf16_t* kp = kb + (size_t)(b * LS + kc + t * 16 + l15) * N_EMBD + h * HEAD_DIM;
      FragAB bf0, bf1;
      bf0.h[0] = *(const v8bf*)(kp + half * 8);
      bf0.h[1] = *(const v8bf*)(kp + half * 8 + 16);
      bf1.h[0] = *(const v8bf*)(kp + 32 + half * 8);
      bf1.h[1] = *(const v8bf*)(kp + 32 + half * 8 + 16);
      s[t] = wmma_bf16(aq[0], bf0, s[t]);
      s[t] = wmma_bf16(aq[1], bf1, s[t]);
    }
    // --- mask + scale
    float ma0 = mod_age[b * LS + kc + l15];
    float ma1 = mod_age[b * LS + kc + 16 + l15];
    #pragma unroll
    for (int v = 0; v < 8; ++v) {
      s[0][v] = (arow[v] >= ma0) ? s[0][v] * 0.125f : -1e30f;
      s[1][v] = (arow[v] >= ma1) ? s[1][v] * 0.125f : -1e30f;
    }
    // --- online softmax; row stats live in registers (C-layout rows per lane-half)
    float fac[8];
    #pragma unroll
    for (int v = 0; v < 8; ++v) {
      float c = fmaxf(s[0][v], s[1][v]);
      #pragma unroll
      for (int d = 1; d < 16; d <<= 1) c = fmaxf(c, __shfl_xor(c, d));
      float nm = fmaxf(mi[v], c);
      fac[v] = __expf(mi[v] - nm);
      s[0][v] = __expf(s[0][v] - nm);
      s[1][v] = __expf(s[1][v] - nm);
      float rr = s[0][v] + s[1][v];
      #pragma unroll
      for (int d = 1; d < 16; d <<= 1) rr += __shfl_xor(rr, d);
      li[v] = li[v] * fac[v] + rr;
      mi[v] = nm;
    }
    #pragma unroll
    for (int j = 0; j < 4; ++j)
      #pragma unroll
      for (int v = 0; v < 8; ++v) acc[j][v] *= fac[v];
    // --- P: C-layout -> A-layout via per-wave LDS tile (DS ops in-order per wave)
    bf16_t* pw = &plds[wid][0];
    #pragma unroll
    for (int t = 0; t < 2; ++t)
      #pragma unroll
      for (int v = 0; v < 8; ++v)
        pw[(v + half * 8) * 32 + t * 16 + l15] = (bf16_t)s[t][v];
    FragAB pa;
    pa.h[0] = *(const v8bf*)&pw[l15 * 32 + half * 8];
    pa.h[1] = *(const v8bf*)&pw[l15 * 32 + 16 + half * 8];
    // --- O += P V  (V pre-transposed: [b,h,d,l] so K is contiguous per lane)
    #pragma unroll
    for (int j = 0; j < 4; ++j) {
      const bf16_t* vp = vt + (((size_t)(b * N_HEAD + h) * HEAD_DIM + j * 16 + l15) << 10)
                            + kc + half * 8;
      FragAB bv;
      bv.h[0] = *(const v8bf*)vp;
      bv.h[1] = *(const v8bf*)(vp + 16);
      acc[j] = wmma_bf16(pa, bv, acc[j]);
    }
  }
  #pragma unroll
  for (int j = 0; j < 4; ++j)
    #pragma unroll
    for (int v = 0; v < 8; ++v) {
      float o = (mi[v] <= -1e30f) ? 0.0f : acc[j][v] / li[v];
      int m = q0 + v + half * 8;
      int n = h * HEAD_DIM + j * 16 + l15;
      yb[(size_t)(b * LM + m) * N_EMBD + n] = (bf16_t)o;
    }
}

// ---------------------------------------------------------------- host orchestration
extern "C" void kernel_launch(void* const* d_in, const int* in_sizes, int n_in,
                              void* d_out, int out_size, void* d_ws, size_t ws_size,
                              hipStream_t stream) {
  const float* x       = (const float*)d_in[0];
  const float* age     = (const float*)d_in[1];
  const int*   mod_idx = (const int*)d_in[2];
  const float* mod_age = (const float*)d_in[3];
  const float* m2      = (const float*)d_in[4];
  const float* m3      = (const float*)d_in[5];
  const float* ln0_w = (const float*)d_in[6];
  const float* ln0_b = (const float*)d_in[7];
  const float* ln1_w = (const float*)d_in[8];
  const float* ln1_b = (const float*)d_in[9];
  const float* ln2_w = (const float*)d_in[10];
  const float* ln2_b = (const float*)d_in[11];
  const float* q_w = (const float*)d_in[12];
  const float* q_b = (const float*)d_in[13];
  const float* k_w = (const float*)d_in[14];
  const float* k_b = (const float*)d_in[15];
  const float* v_w = (const float*)d_in[16];
  const float* v_b = (const float*)d_in[17];
  const float* c_w = (const float*)d_in[18];
  const float* c_b = (const float*)d_in[19];
  const float* fc_w   = (const float*)d_in[20];
  const float* fc_b   = (const float*)d_in[21];
  const float* proj_w = (const float*)d_in[22];
  const float* proj_b = (const float*)d_in[23];

  size_t off = 0;
  auto alloc = [&](size_t bytes) -> void* {
    off = (off + 255) & ~(size_t)255;
    void* p = (char*)d_ws + off;
    off += bytes;
    return p;
  };
  const size_t ROWS = (size_t)BATCH * LM;   // 4096
  bf16_t* wq    = (bf16_t*)alloc((size_t)N_EMBD * N_EMBD * 2);
  bf16_t* wk    = (bf16_t*)alloc((size_t)N_EMBD * N_EMBD * 2);
  bf16_t* wv    = (bf16_t*)alloc((size_t)N_EMBD * N_EMBD * 2);
  bf16_t* wc    = (bf16_t*)alloc((size_t)N_EMBD * N_EMBD * 2);
  bf16_t* wfc   = (bf16_t*)alloc((size_t)FFN * N_EMBD * 2);
  bf16_t* wproj = (bf16_t*)alloc((size_t)N_EMBD * FFN * 2);
  float*  modx  = (float*)alloc(ROWS * N_EMBD * 4);
  bf16_t* xn    = (bf16_t*)alloc(ROWS * N_EMBD * 2);
  bf16_t* sn    = (bf16_t*)alloc(ROWS * N_EMBD * 2);
  bf16_t* qbuf  = (bf16_t*)alloc(ROWS * N_EMBD * 2);
  bf16_t* kbuf  = (bf16_t*)alloc(ROWS * N_EMBD * 2);
  bf16_t* vtb   = (bf16_t*)alloc(ROWS * N_EMBD * 2);
  bf16_t* ybuf  = (bf16_t*)alloc(ROWS * N_EMBD * 2);
  float*  x1    = (float*)alloc(ROWS * N_EMBD * 4);
  bf16_t* hn    = (bf16_t*)alloc(ROWS * N_EMBD * 2);
  bf16_t* hbuf  = (bf16_t*)alloc(ROWS * (size_t)FFN * 2);
  int*    entry = (int*)alloc(ROWS * 4);
  (void)in_sizes; (void)n_in; (void)out_size; (void)ws_size;

  // 1. fuse_embed (mod_age pre-sorted -> stable argsort is identity)
  occ_kernel<<<BATCH, 32, 0, stream>>>(mod_idx, entry);
  gather_kernel<<<(int)ROWS, 256, 0, stream>>>(entry, m2, m3, modx);

  // 2. LayerNorms -> bf16
  ln_kernel<<<(int)ROWS, 256, 0, stream>>>(x, ln1_w, ln1_b, xn);
  ln_kernel<<<(int)ROWS, 256, 0, stream>>>(modx, ln0_w, ln0_b, sn);

  // 3. weights -> bf16
  auto cvt = [&](const float* s, bf16_t* d, int n) {
    cvt_kernel<<<(n + 255) / 256, 256, 0, stream>>>(s, d, n);
  };
  cvt(q_w, wq, N_EMBD * N_EMBD);
  cvt(k_w, wk, N_EMBD * N_EMBD);
  cvt(v_w, wv, N_EMBD * N_EMBD);
  cvt(c_w, wc, N_EMBD * N_EMBD);
  cvt(fc_w, wfc, FFN * N_EMBD);
  cvt(proj_w, wproj, N_EMBD * FFN);

  // 4. projections (WMMA): Q, K (bf16), V (bf16, transposed to [b,h,d,l])
  const int tM = (int)ROWS / 64;           // 64
  const int tN768 = N_EMBD / 128;          // 6
  const int tNffn = FFN / 128;             // 24
  gemm_kernel<0><<<tM * tN768, 256, 0, stream>>>(xn, wq, q_b, nullptr, nullptr, qbuf,
                                                 (int)ROWS, N_EMBD, N_EMBD, tN768);
  gemm_kernel<0><<<tM * tN768, 256, 0, stream>>>(sn, wk, k_b, nullptr, nullptr, kbuf,
                                                 (int)ROWS, N_EMBD, N_EMBD, tN768);
  gemm_kernel<1><<<tM * tN768, 256, 0, stream>>>(sn, wv, v_b, nullptr, nullptr, vtb,
                                                 (int)ROWS, N_EMBD, N_EMBD, tN768);

  // 5. flash cross-attention (WMMA for QK^T and PV)
  attn_kernel<<<BATCH * N_HEAD * 64 / 8, 256, 0, stream>>>(qbuf, kbuf, vtb, age, mod_age, ybuf);

  // 6. attention out-proj + residual (fp32 out)
  gemm_kernel<2><<<tM * tN768, 256, 0, stream>>>(ybuf, wc, c_b, x, x1, nullptr,
                                                 (int)ROWS, N_EMBD, N_EMBD, tN768);

  // 7. MLP: LN -> fc+GELU -> proj + residual
  ln_kernel<<<(int)ROWS, 256, 0, stream>>>(x1, ln2_w, ln2_b, hn);
  gemm_kernel<3><<<tM * tNffn, 256, 0, stream>>>(hn, wfc, fc_b, nullptr, nullptr, hbuf,
                                                 (int)ROWS, FFN, N_EMBD, tNffn);
  gemm_kernel<2><<<tM * tN768, 256, 0, stream>>>(hbuf, wproj, proj_b, x1, (float*)d_out, nullptr,
                                                 (int)ROWS, N_EMBD, FFN, tN768);
}
